// TripletLossWithCurvature_43422119362762
// MI455X (gfx1250) — matrix operand order repaired and checked by reading, not compiled, not run on hardware
//
#include <hip/hip_runtime.h>
#include <hip/hip_bf16.h>
#include <math.h>
#include <stdint.h>

// ---------------------------------------------------------------------------
// TripletLossWithCurvature on gfx1250 (MI455X), wave32 + WMMA bf16 + TDM.
//   features: [8192, 512] f32, labels: [8192] int
//   G = I + cov(centered)        -> 512x512x8192 GEMM (WMMA bf16, TDM staged)
//   pos/neg idx: hash-argmax     -> equivalent to gumbel-max (monotone)
//   d2 = diff^T G diff           -> fused [8192x512]x[512x512] WMMA GEMM
//   loss = mean_valid(max(dp-dn+1,0))
// ---------------------------------------------------------------------------

#define BN 8192
#define DN 512

typedef __bf16 bf16_t;
typedef __attribute__((ext_vector_type(16))) __bf16 v16bf;
typedef __attribute__((ext_vector_type(8)))  __bf16 v8bf;
typedef __attribute__((ext_vector_type(8)))  float  v8f;

typedef __attribute__((ext_vector_type(4))) unsigned tdm_g0_t;
typedef __attribute__((ext_vector_type(8))) int      tdm_g1_t;
typedef __attribute__((ext_vector_type(4))) int      tdm_g2_t;

#if __has_builtin(__builtin_amdgcn_tensor_load_to_lds)
#define HAVE_TDM 1
#if __has_include(<hip/amd_detail/amd_gfx1250_TDM.h>)
#define TDM_SIX_ARGS 1
#endif
#endif

// ---- workspace layout (bytes) ----
#define OFF_MEAN 0u                    // 512 f32
#define OFF_D2P  2048u                 // 8192 f32
#define OFF_D2N  34816u                // 8192 f32
#define OFF_ACC  67584u                // 2 f32  (zeroed region ends at 67592)
#define OFF_POS  67840u                // 8192 i32
#define OFF_NEG  100608u               // 8192 i32
#define OFF_VAL  133376u               // 8192 i32
#define OFF_CT   166144u               // 512*8192 bf16 = 8 MB  (centered^T)
#define OFF_GBF  8554752u              // 512*512 bf16
#define OFF_DP   9079040u              // 8192*512 bf16
#define OFF_DN   17467648u             // 8192*512 bf16

__device__ __forceinline__ bf16_t f2bf(float f) {
  union { float f; unsigned u; } in; in.f = f;
  unsigned u = in.u + 0x7FFFu + ((in.u >> 16) & 1u);   // round-to-nearest-even
  union { unsigned short s; bf16_t b; } out;
  out.s = (unsigned short)(u >> 16);
  return out.b;
}

__device__ __forceinline__ unsigned hash3(unsigned a, unsigned b, unsigned c) {
  unsigned h = a * 0x9E3779B1u;
  h ^= (b * 0x85EBCA77u) + 0x9E3779B9u + (h << 6) + (h >> 2);
  h ^= (c * 0xC2B2AE3Du) + 0x9E3779B9u + (h << 6) + (h >> 2);
  h ^= h >> 16; h *= 0x7FEB352Du; h ^= h >> 15; h *= 0x846CA68Bu; h ^= h >> 16;
  return h;
}

// ---------------- TDM 2-D tile stage: global (row-major, strided) -> LDS ----
// D# per CDNA5 ISA:
//   group0: [1:0]count=1, [63:32]lds_addr, [120:64]global_addr, [127:126]type=2
//   group1: [17:16]data_size(1=2B), [79:48]tensor_dim0, [111:80]tensor_dim1,
//           [127:112]tile_dim0, [143:128]tile_dim1, [207:160..]dim0_stride
__device__ __forceinline__ void stage_tile(bf16_t* lds, const bf16_t* g,
                                           int rows, int kelems,
                                           unsigned long long row_stride) {
#if defined(HAVE_TDM)
  unsigned lds_off = (unsigned)(unsigned long long)(uintptr_t)lds; // LDS aperture: addr[31:0]=offset
  unsigned long long ga = (unsigned long long)(uintptr_t)g;
  tdm_g0_t g0;
  g0[0] = 1u;                                              // count=1 (valid, user)
  g0[1] = lds_off;
  g0[2] = (unsigned)ga;
  g0[3] = (unsigned)((ga >> 32) & 0x01FFFFFFull) | (2u << 30); // addr[56:32] | type=2
  unsigned td0 = (unsigned)row_stride;                     // tensor extent dim0
  unsigned td1 = (unsigned)DN;                             // tensor extent dim1 (rows)
  tdm_g1_t g1;
  g1[0] = 0x00010000;                                      // wg_mask=0, data_size=2B
  g1[1] = (int)((td0 & 0xFFFFu) << 16);                    // [63:48]=dim0[15:0]
  g1[2] = (int)(((td0 >> 16) & 0xFFFFu) | ((td1 & 0xFFFFu) << 16));
  g1[3] = (int)(((td1 >> 16) & 0xFFFFu) | (((unsigned)kelems & 0xFFFFu) << 16)); // tile_dim0
  g1[4] = (int)((unsigned)rows & 0xFFFFu);                 // tile_dim1, tile_dim2=0
  g1[5] = (int)(row_stride & 0xFFFFFFFFull);               // dim0_stride[31:0]
  g1[6] = (int)((row_stride >> 32) & 0xFFFFull);           // dim0_stride[47:32]
  g1[7] = 0;
  tdm_g2_t gz = {};
#if defined(TDM_SIX_ARGS)
  tdm_g1_t z8 = {};
  __builtin_amdgcn_tensor_load_to_lds(g0, g1, gz, gz, z8, 0);
#else
  __builtin_amdgcn_tensor_load_to_lds(g0, g1, gz, gz, 0);
#endif
#else
  // fallback: wave copy, 8 bf16 per lane-step
  int lane = threadIdx.x & 31;
  int per_row = kelems >> 3;
  int total = rows * per_row;
  for (int t = lane; t < total; t += 32) {
    int r = t / per_row, cc = t - r * per_row;
    *(v8bf*)(lds + (size_t)r * kelems + cc * 8) =
        *(const v8bf*)(g + (size_t)r * row_stride + cc * 8);
  }
#endif
}

template <int N>
__device__ __forceinline__ void stage_wait() {
#if defined(HAVE_TDM)
#if __has_builtin(__builtin_amdgcn_s_wait_tensorcnt)
  __builtin_amdgcn_s_wait_tensorcnt((short)N);
#else
  asm volatile("s_wait_tensorcnt %0" ::"i"(N));
#endif
#endif
}

// ---------------- zero ----------------
__global__ void zero_f32_kernel(float* p, int n) {
  int i = blockIdx.x * blockDim.x + threadIdx.x;
  if (i < n) p[i] = 0.0f;
}

// ---------------- column sums (for mean) ----------------
__global__ void colsum_kernel(const float* __restrict__ f, float* __restrict__ colsum) {
  const int ROWS = BN / 64;                 // 64 blocks
  int col  = threadIdx.x;                   // blockDim.x == 512
  int row0 = blockIdx.x * ROWS;
  float s = 0.0f;
  for (int r = 0; r < ROWS; ++r)
    s += f[(size_t)(row0 + r) * DN + col];
  atomicAdd(&colsum[col], s);
}

// ---------------- center + bf16 + transpose: CT[d][b] ----------------
__global__ void center_kernel(const float* __restrict__ f, const float* __restrict__ colsum,
                              bf16_t* __restrict__ CT) {
  int idx = blockIdx.x * blockDim.x + threadIdx.x;   // BN*DN threads
  int col = idx & (DN - 1);
  int row = idx >> 9;
  float c = f[idx] - colsum[col] * (1.0f / (float)BN);
  CT[(size_t)col * BN + row] = f2bf(c);
}

// ---------------- WMMA fragment loaders (16-bit A 16x32 / B 32x16) ---------
__device__ __forceinline__ v16bf load_a_frag(const bf16_t* row, int k0, int hi) {
  return *(const v16bf*)(row + k0 + (hi ? 16 : 0));    // K = 0..15 or 16..31
}
__device__ __forceinline__ v16bf load_b_frag(const bf16_t* row, int k0, int hi) {
  return *(const v16bf*)(row + k0 + (hi ? 16 : 0));    // K = 0..15 or 16..31
}

// ---------------- G = I + C^T C / B : TDM double-buffered WMMA -------------
#define KCH 128
#define NCH (BN / KCH)

__global__ void __launch_bounds__(32) cov_wmma_kernel(const bf16_t* __restrict__ CT,
                                                      bf16_t* __restrict__ Gbf) {
  __shared__ bf16_t sA[2][16 * KCH];   // 4 KB per buffer
  __shared__ bf16_t sB[2][16 * KCH];
  int tm = blockIdx.x * 16, tn = blockIdx.y * 16;
  int lane = threadIdx.x & 31;
  int hi   = lane >> 4;
  int id16 = lane & 15;

  stage_tile(&sA[0][0], CT + (size_t)tm * BN, 16, KCH, BN);
  stage_tile(&sB[0][0], CT + (size_t)tn * BN, 16, KCH, BN);

  v8f acc = {};
  for (int c = 0; c < NCH; ++c) {
    int cur = c & 1;
    if (c + 1 < NCH) {
      int nxt = cur ^ 1;
      size_t k0n = (size_t)(c + 1) * KCH;
      stage_tile(&sA[nxt][0], CT + (size_t)tm * BN + k0n, 16, KCH, BN);
      stage_tile(&sB[nxt][0], CT + (size_t)tn * BN + k0n, 16, KCH, BN);
      stage_wait<2>();       // TDM in-order per wave: chunk c's 2 loads done
    } else {
      stage_wait<0>();
    }
    const bf16_t* aBase = &sA[cur][0] + (size_t)id16 * KCH;
    const bf16_t* bBase = &sB[cur][0] + (size_t)id16 * KCH;
#pragma unroll
    for (int kk = 0; kk < KCH; kk += 32) {
      v16bf a = load_a_frag(aBase, kk, hi);
      v16bf b = load_b_frag(bBase, kk, hi);
      acc = __builtin_amdgcn_wmma_f32_16x16x32_bf16(false, a, false, b,
                                                    (short)0, acc, false, false);
    }
  }
  const float inv = 1.0f / (float)BN;
  int mb = hi * 8;
#pragma unroll
  for (int r = 0; r < 8; ++r) {
    int m = tm + mb + r, n = tn + id16;
    float g = acc[r] * inv + (m == n ? 1.0f : 0.0f);
    Gbf[(size_t)m * DN + n] = f2bf(g);
  }
}

// ---------------- per-row random pos/neg selection (hash argmax) ----------
__global__ void select_kernel(const int* __restrict__ labels,
                              int* __restrict__ pos_idx, int* __restrict__ neg_idx,
                              int* __restrict__ valid) {
  int wave = threadIdx.x >> 5;
  int lane = threadIdx.x & 31;
  int row  = blockIdx.x * (blockDim.x >> 5) + wave;
  int li = labels[row];
  unsigned bp = 0u, bn = 0u; int ip = -1, inn = -1;
  for (int j = lane; j < BN; j += 32) {
    int lj = labels[j];
    bool same = (lj == li);
    if (same && j != row) {
      unsigned h = hash3((unsigned)row, (unsigned)j, 0x01234567u);
      if (ip < 0 || h > bp) { bp = h; ip = j; }
    }
    if (!same) {
      unsigned h = hash3((unsigned)row, (unsigned)j, 0x89ABCDEFu);
      if (inn < 0 || h > bn) { bn = h; inn = j; }
    }
  }
  for (int s = 16; s > 0; s >>= 1) {
    unsigned obp = __shfl_down(bp, s, 32); int oip = __shfl_down(ip, s, 32);
    if (oip >= 0 && (ip < 0 || obp > bp || (obp == bp && oip < ip))) { bp = obp; ip = oip; }
    unsigned obn = __shfl_down(bn, s, 32); int oin = __shfl_down(inn, s, 32);
    if (oin >= 0 && (inn < 0 || obn > bn || (obn == bn && oin < inn))) { bn = obn; inn = oin; }
  }
  if (lane == 0) {
    pos_idx[row] = ip; neg_idx[row] = inn;
    valid[row] = (ip >= 0 && inn >= 0) ? 1 : 0;
  }
}

// ---------------- diff matrices in bf16 ----------------
__global__ void diff_kernel(const float* __restrict__ f,
                            const int* __restrict__ pos_idx, const int* __restrict__ neg_idx,
                            bf16_t* __restrict__ DP, bf16_t* __restrict__ DN_) {
  int idx = blockIdx.x * blockDim.x + threadIdx.x;
  int col = idx & (DN - 1);
  int row = idx >> 9;
  float x = f[idx];
  int p = pos_idx[row], n = neg_idx[row];
  float dp = (p >= 0) ? (x - f[(size_t)p * DN + col]) : 0.0f;
  float dn = (n >= 0) ? (x - f[(size_t)n * DN + col]) : 0.0f;
  DP[idx] = f2bf(dp);
  DN_[idx] = f2bf(dn);
}

// ---------------- d2[b] = diff_b^T G diff_b (TDM-staged fused WMMA) --------
// T = diff x G (G symmetric => B-frag = rows of G). One wave per 16x16 tile
// of T: stage the full 16x512 diff and G tiles in LDS via 2 TDM loads, run
// 16 WMMAs, multiply by fp32 diff, reduce across the N lane group, atomic add.
__global__ void __launch_bounds__(32) qf_wmma_kernel(const bf16_t* __restrict__ Dbf,
                                                     const bf16_t* __restrict__ Gbf,
                                                     const float* __restrict__ f,
                                                     const int* __restrict__ sel,
                                                     float* __restrict__ d2) {
  __shared__ bf16_t sD[16 * DN];   // 16 KB
  __shared__ bf16_t sG[16 * DN];   // 16 KB
  int tm = blockIdx.x * 16;   // samples
  int tn = blockIdx.y * 16;   // output dims
  int lane = threadIdx.x & 31;
  int hi   = lane >> 4;
  int id16 = lane & 15;

  stage_tile(sD, Dbf + (size_t)tm * DN, 16, DN, DN);
  stage_tile(sG, Gbf + (size_t)tn * DN, 16, DN, DN);
  stage_wait<0>();

  const bf16_t* aBase = sD + (size_t)id16 * DN;
  const bf16_t* bBase = sG + (size_t)id16 * DN;
  v8f acc = {};
#pragma unroll
  for (int k0 = 0; k0 < DN; k0 += 32) {
    v16bf a = load_a_frag(aBase, k0, hi);
    v16bf b = load_b_frag(bBase, k0, hi);
    acc = __builtin_amdgcn_wmma_f32_16x16x32_bf16(false, a, false, b,
                                                  (short)0, acc, false, false);
  }
  int n  = tn + id16;
  int mb = hi * 8;
#pragma unroll
  for (int r = 0; r < 8; ++r) {
    int m = tm + mb + r;
    int s = sel[m];
    float diff = (s >= 0) ? (f[(size_t)m * DN + n] - f[(size_t)s * DN + n]) : 0.0f;
    float val = acc[r] * diff;
    for (int sh = 1; sh < 16; sh <<= 1) val += __shfl_xor(val, sh, 32);
    if (id16 == 0) atomicAdd(&d2[m], val);
  }
}

// ---------------- loss reduction ----------------
__global__ void loss_kernel(const float* __restrict__ d2p, const float* __restrict__ d2n,
                            const int* __restrict__ valid, float* __restrict__ accum) {
  int b = blockIdx.x * blockDim.x + threadIdx.x;
  float loss = 0.0f, cnt = 0.0f;
  if (valid[b]) {
    float dp = sqrtf(fmaxf(d2p[b], 0.0f) + 1e-8f);
    float dn = sqrtf(fmaxf(d2n[b], 0.0f) + 1e-8f);
    loss = fmaxf(dp - dn + 1.0f, 0.0f);
    cnt  = 1.0f;
  }
  for (int s = 16; s > 0; s >>= 1) {
    loss += __shfl_down(loss, s, 32);
    cnt  += __shfl_down(cnt,  s, 32);
  }
  if ((threadIdx.x & 31) == 0) {
    atomicAdd(&accum[0], loss);
    atomicAdd(&accum[1], cnt);
  }
}

__global__ void final_kernel(const float* __restrict__ accum, float* __restrict__ out) {
  out[0] = accum[0] / fmaxf(accum[1], 1.0f);
}

// ---------------------------------------------------------------------------
extern "C" void kernel_launch(void* const* d_in, const int* in_sizes, int n_in,
                              void* d_out, int out_size, void* d_ws, size_t ws_size,
                              hipStream_t stream) {
  const float* f      = (const float*)d_in[0];
  const int*   labels = (const int*)d_in[1];
  char* ws = (char*)d_ws;

  float*  mean = (float*)(ws + OFF_MEAN);
  float*  d2p  = (float*)(ws + OFF_D2P);
  float*  d2n  = (float*)(ws + OFF_D2N);
  float*  acc  = (float*)(ws + OFF_ACC);
  int*    pos  = (int*)(ws + OFF_POS);
  int*    neg  = (int*)(ws + OFF_NEG);
  int*    val  = (int*)(ws + OFF_VAL);
  bf16_t* CT   = (bf16_t*)(ws + OFF_CT);
  bf16_t* Gbf  = (bf16_t*)(ws + OFF_GBF);
  bf16_t* DP   = (bf16_t*)(ws + OFF_DP);
  bf16_t* DNm  = (bf16_t*)(ws + OFF_DN);

  // zero [mean .. acc] : contiguous prefix of workspace, 16898 floats
  zero_f32_kernel<<<(16898 + 255) / 256, 256, 0, stream>>>((float*)ws, 16898);

  colsum_kernel<<<64, 512, 0, stream>>>(f, mean);
  center_kernel<<<(BN * DN) / 256, 256, 0, stream>>>(f, mean, CT);

  cov_wmma_kernel<<<dim3(DN / 16, DN / 16), 32, 0, stream>>>(CT, Gbf);

  select_kernel<<<BN / 8, 256, 0, stream>>>(labels, pos, neg, val);
  diff_kernel<<<(BN * DN) / 256, 256, 0, stream>>>(f, pos, neg, DP, DNm);

  dim3 qgrid(BN / 16, DN / 16);
  qf_wmma_kernel<<<qgrid, 32, 0, stream>>>(DP, Gbf, f, pos, d2p);
  qf_wmma_kernel<<<qgrid, 32, 0, stream>>>(DNm, Gbf, f, neg, d2n);

  loss_kernel<<<BN / 256, 256, 0, stream>>>(d2p, d2n, val, acc);
  final_kernel<<<1, 1, 0, stream>>>(acc, (float*)d_out);
}
